// PeerHTC_33139967656303
// MI455X (gfx1250) — matrix operand order repaired and checked by reading, not compiled, not run on hardware
//
#include <hip/hip_runtime.h>
#include <hip/hip_bf16.h>
#include <math.h>

typedef __bf16 bf16_t;
typedef __attribute__((ext_vector_type(16))) __bf16 v16bf;
typedef __attribute__((ext_vector_type(8)))  __bf16 v8bf;
typedef __attribute__((ext_vector_type(8)))  float  v8f;
typedef __attribute__((ext_vector_type(4)))  float  v4f;

// ---------------------------------------------------------------------------
// Problem constants
// ---------------------------------------------------------------------------
#define HH   256
#define LL   141
#define NN1  9
#define NN2  34
#define NN3  98
#define BB   64
#define SS   512
#define LPAD 144          // 141 padded to 9 WMMA tiles
#define KOUT 36096        // 141*256
#define ACT_SIG  0
#define ACT_TANH 1
#define ACT_RELU 2
#define ACT_NONE 3

// ===========================================================================
// WMMA fragment helpers (layouts used by this code, wave32)
//   A 16x32 bf16: lane m = lane&15, K base = (lane>>4)*16;
//                 elements 0-15 -> K=base+e (contiguous, same as B)
//   B 32x16 bf16: lane n = lane&15, K = (lane>>4)*16 + e (contiguous)
//   C 16x16 f32 : row = r + (lane>>4)*8, col = lane&15
// ===========================================================================
__device__ __forceinline__ v16bf frag_a_bf(const bf16_t* __restrict__ p) {
    // p = &src[row*ld + k0 + (lane>>4)*16]; 16-byte aligned
    v8bf lo8 = *(const v8bf*)p;
    v8bf hi8 = *(const v8bf*)(p + 8);
    return __builtin_shufflevector(lo8, hi8, 0,1,2,3,4,5,6,7,8,9,10,11,12,13,14,15);
}
__device__ __forceinline__ v16bf frag_a_f32(const float* __restrict__ p) {
    v4f x0 = *(const v4f*)(p);
    v4f x1 = *(const v4f*)(p + 4);
    v4f x2 = *(const v4f*)(p + 8);
    v4f x3 = *(const v4f*)(p + 12);
    v16bf a = {};
#pragma unroll
    for (int e = 0; e < 4; ++e) {
        a[e]      = (bf16_t)x0[e];
        a[e + 4]  = (bf16_t)x1[e];
        a[e + 8]  = (bf16_t)x2[e];
        a[e + 12] = (bf16_t)x3[e];
    }
    return a;
}
__device__ __forceinline__ v16bf frag_b_contig(const bf16_t* __restrict__ q) {
    // q = &B[(col n)*ldk + k0 + (lane>>4)*16]; 32-byte aligned
    return *(const v16bf*)q;
}
#define WMMA_BF16(a, b, c) \
    __builtin_amdgcn_wmma_f32_16x16x32_bf16(false, (a), false, (b), (short)0, (c), false, false)

// ===========================================================================
// Small VALU kernels for the label-encoding LSTM chain (tiny FLOPs)
// ===========================================================================
__global__ void affine_act_kernel(const float* __restrict__ X, int K,
                                  const float* __restrict__ W,
                                  const float* __restrict__ bias,
                                  const float* __restrict__ Y,
                                  const float* __restrict__ U,
                                  float* __restrict__ out, int M, int act) {
    int idx = blockIdx.x * blockDim.x + threadIdx.x;
    if (idx >= M * HH) return;
    int r = idx >> 8, c = idx & 255;
    float s = bias ? bias[c] : 0.f;
    const float* xr = X + (size_t)r * K;
    for (int k = 0; k < K; ++k) s += xr[k] * W[k * HH + c];
    if (Y) {
        const float* yr = Y + (size_t)r * HH;
        for (int k = 0; k < HH; ++k) s += yr[k] * U[k * HH + c];
    }
    if (act == ACT_SIG)       s = 1.f / (1.f + expf(-s));
    else if (act == ACT_TANH) s = tanhf(s);
    else if (act == ACT_RELU) s = fmaxf(s, 0.f);
    out[idx] = s;
}

__global__ void mm_kernel(const float* __restrict__ A, const float* __restrict__ B,
                          float* __restrict__ C, int M, int K, int lda, int transA) {
    int idx = blockIdx.x * blockDim.x + threadIdx.x;
    if (idx >= M * HH) return;
    int r = idx >> 8, c = idx & 255;
    float s = 0.f;
    if (transA) for (int k = 0; k < K; ++k) s += A[k * lda + r] * B[k * HH + c];
    else        for (int k = 0; k < K; ++k) s += A[r * lda + k] * B[k * HH + c];
    C[idx] = s;
}

__global__ void combine_kernel(const float* __restrict__ gi, const float* __restrict__ gu,
                               const float* __restrict__ go, const float* __restrict__ gf,
                               const float* __restrict__ cin, const float* __restrict__ add,
                               float* __restrict__ h, float* __restrict__ cout, int M) {
    int idx = blockIdx.x * blockDim.x + threadIdx.x;
    if (idx >= M * HH) return;
    float c = gi[idx] * gu[idx];
    if (gf)  c += gf[idx] * cin[idx];
    if (add) c += add[idx];
    if (cout) cout[idx] = c;
    h[idx] = go[idx] * tanhf(c);
}

__global__ void emul_kernel(const float* a, const float* b, float* t, int n) {
    int idx = blockIdx.x * blockDim.x + threadIdx.x;
    if (idx < n) t[idx] = a[idx] * b[idx];
}

__global__ void zero_kernel(float* p, int n) {
    int idx = blockIdx.x * blockDim.x + threadIdx.x;
    if (idx < n) p[idx] = 0.f;
}

__global__ void assemble_kernel(const float* h11, const float* h12,
                                const float* h21, const float* h22,
                                const float* h31, const float* h32,
                                const float* temp, float* le_cat) {
    int idx = blockIdx.x * blockDim.x + threadIdx.x;
    if (idx >= LL * 768) return;
    int r = idx / 768, c = idx % 768;
    float v;
    if (c < 256) {
        if (r < NN1)            v = h11[r * HH + c];
        else if (r < NN1 + NN2) v = h21[(r - NN1) * HH + c];
        else                    v = h31[(r - NN1 - NN2) * HH + c];
    } else if (c < 512) {
        int cc = c - 256;
        if (r < NN1)            v = h12[r * HH + cc];
        else if (r < NN1 + NN2) v = h22[(r - NN1) * HH + cc];
        else                    v = h32[(r - NN1 - NN2) * HH + cc];
    } else {
        v = temp[r * HH + (c - 512)];
    }
    le_cat[idx] = v;
}

__global__ void conv_le_kernel(const float* __restrict__ le, bf16_t* __restrict__ le_bf) {
    int idx = blockIdx.x * blockDim.x + threadIdx.x;
    if (idx >= LPAD * HH) return;
    int r = idx >> 8;
    le_bf[idx] = (r < LL) ? (bf16_t)le[idx] : (bf16_t)0.f;
}

// tt_W (768x256 f32) -> N-major bf16: ttw_t[n*768+k]
__global__ void convttw_kernel(const float* __restrict__ W, bf16_t* __restrict__ Wt) {
    int idx = blockIdx.x * blockDim.x + threadIdx.x;
    if (idx >= 768 * HH) return;
    int k = idx >> 8, n = idx & 255;
    Wt[(size_t)n * 768 + k] = (bf16_t)W[idx];
}

// out_W (36096x141 f32) -> N-major bf16 padded: outw_t[j*36096+k], j<144
__global__ void convw_kernel(const float* __restrict__ W, bf16_t* __restrict__ Wt) {
    long idx = (long)blockIdx.x * blockDim.x + threadIdx.x;
    if (idx >= (long)LPAD * KOUT) return;
    int j = (int)(idx / KOUT), k = (int)(idx % KOUT);
    Wt[idx] = (j < LL) ? (bf16_t)W[(size_t)k * LL + j] : (bf16_t)0.f;
}

// ===========================================================================
// WMMA kernel 1: text = tanh(text_hidden @ tt_W + tt_b)
//   M = 32768, K = 768, N = 256; writes bf16 in (s,h) and (h,s) layouts
// ===========================================================================
__global__ void __launch_bounds__(256)
text_kernel(const float* __restrict__ TH, const bf16_t* __restrict__ ttw_t,
            const float* __restrict__ ttb,
            bf16_t* __restrict__ text_bf, bf16_t* __restrict__ text_tbf) {
    int wave = threadIdx.x >> 5, lane = threadIdx.x & 31;
    int hi = lane >> 4, lo = lane & 15;
    int row0 = blockIdx.x * 16;                 // global row (b*512+s) tile
    int n0 = wave * 32;                         // N cols n0..n0+31
    const float*  arow = TH + (size_t)(row0 + lo) * 768 + hi * 16;
    const bf16_t* b0p  = ttw_t + (size_t)(n0 + lo) * 768 + hi * 16;
    const bf16_t* b1p  = ttw_t + (size_t)(n0 + 16 + lo) * 768 + hi * 16;
    v8f c0 = {}, c1 = {};
    for (int k0 = 0; k0 < 768; k0 += 32) {
        v16bf a  = frag_a_f32(arow + k0);
        v16bf b0 = frag_b_contig(b0p + k0);
        v16bf b1 = frag_b_contig(b1p + k0);
        c0 = WMMA_BF16(a, b0, c0);
        c1 = WMMA_BF16(a, b1, c1);
    }
#pragma unroll
    for (int r = 0; r < 8; ++r) {
        int m = row0 + r + hi * 8;
        int b = m >> 9, s = m & 511;
        int n = n0 + lo;
        float v0 = tanhf(c0[r] + ttb[n]);
        float v1 = tanhf(c1[r] + ttb[n + 16]);
        text_bf[(size_t)m * HH + n]      = (bf16_t)v0;
        text_bf[(size_t)m * HH + n + 16] = (bf16_t)v1;
        text_tbf[(size_t)b * (HH * SS) + (size_t)n * SS + s]        = (bf16_t)v0;
        text_tbf[(size_t)b * (HH * SS) + (size_t)(n + 16) * SS + s] = (bf16_t)v1;
    }
}

// ===========================================================================
// WMMA kernel 2: logits[b,l,s] = le[l,:] . text[b,s,:]   (M=144,N=512,K=256)
// ===========================================================================
__global__ void __launch_bounds__(256)
att_logits_kernel(const bf16_t* __restrict__ le_bf, const bf16_t* __restrict__ text_bf,
                  float* __restrict__ logits) {
    int wave = threadIdx.x >> 5, lane = threadIdx.x & 31;
    int hi = lane >> 4, lo = lane & 15;
    int b = blockIdx.x, mt = blockIdx.y;
    int m0 = mt * 16;
    const bf16_t* arow = le_bf + (size_t)(m0 + lo) * HH + hi * 16;
    for (int nt = 0; nt < 4; ++nt) {
        int n0 = (wave * 4 + nt) * 16;          // s tile
        const bf16_t* brow = text_bf + (size_t)(b * SS + n0 + lo) * HH + hi * 16;
        v8f c = {};
        for (int k0 = 0; k0 < HH; k0 += 32) {
            v16bf a  = frag_a_bf(arow + k0);
            v16bf bf = frag_b_contig(brow + k0);
            c = WMMA_BF16(a, bf, c);
        }
#pragma unroll
        for (int r = 0; r < 8; ++r) {
            int m = m0 + r + hi * 8;
            logits[((size_t)b * LPAD + m) * SS + n0 + lo] = c[r];
        }
    }
}

// ===========================================================================
// Masked softmax over s; writes bf16 attention (pad rows l>=141 zeroed)
// ===========================================================================
__global__ void __launch_bounds__(128)
softmax_kernel(const float* __restrict__ logits, const int* __restrict__ tok,
               bf16_t* __restrict__ att_bf) {
    int b = blockIdx.x, l = blockIdx.y, t = threadIdx.x;
    bf16_t* orow = att_bf + ((size_t)b * LPAD + l) * SS;
    if (l >= LL) {
        for (int s = t; s < SS; s += 128) orow[s] = (bf16_t)0.f;
        return;
    }
    const float* row = logits + ((size_t)b * LPAD + l) * SS;
    __shared__ float red[128];
    float x[4]; bool msk[4];
    float mx = -1e30f;
#pragma unroll
    for (int i = 0; i < 4; ++i) {
        int s = t + i * 128;
        int tk = tok[b * SS + s];
        msk[i] = (tk == 0) || (tk == 101) || (tk == 102);
        x[i] = row[s];
        if (!msk[i]) mx = fmaxf(mx, x[i]);
    }
    red[t] = mx; __syncthreads();
    for (int o = 64; o > 0; o >>= 1) { if (t < o) red[t] = fmaxf(red[t], red[t + o]); __syncthreads(); }
    mx = red[0]; __syncthreads();
    float e[4], sm = 0.f;
#pragma unroll
    for (int i = 0; i < 4; ++i) { e[i] = msk[i] ? 0.f : expf(x[i] - mx); sm += e[i]; }
    red[t] = sm; __syncthreads();
    for (int o = 64; o > 0; o >>= 1) { if (t < o) red[t] += red[t + o]; __syncthreads(); }
    float inv = 1.f / red[0];
#pragma unroll
    for (int i = 0; i < 4; ++i) orow[t + i * 128] = (bf16_t)(e[i] * inv);
}

// ===========================================================================
// WMMA kernel 3: features[b,l,h] = sum_s att[b,l,s]*text[b,s,h] (M=144,N=256,K=512)
// ===========================================================================
__global__ void __launch_bounds__(256)
feat_kernel(const bf16_t* __restrict__ att_bf, const bf16_t* __restrict__ text_tbf,
            bf16_t* __restrict__ features_bf) {
    int wave = threadIdx.x >> 5, lane = threadIdx.x & 31;
    int hi = lane >> 4, lo = lane & 15;
    int b = blockIdx.x, mt = blockIdx.y;
    int m0 = mt * 16;
    const bf16_t* arow = att_bf + ((size_t)b * LPAD + m0 + lo) * SS + hi * 16;
    for (int nt = 0; nt < 2; ++nt) {
        int n0 = (wave * 2 + nt) * 16;           // h tile
        const bf16_t* brow = text_tbf + (size_t)b * (HH * SS) + (size_t)(n0 + lo) * SS + hi * 16;
        v8f c = {};
        for (int k0 = 0; k0 < SS; k0 += 32) {
            v16bf a  = frag_a_bf(arow + k0);
            v16bf bf = frag_b_contig(brow + k0);
            c = WMMA_BF16(a, bf, c);
        }
#pragma unroll
        for (int r = 0; r < 8; ++r) {
            int m = m0 + r + hi * 8;
            if (m < LL)
                features_bf[(size_t)b * KOUT + (size_t)m * HH + n0 + lo] = (bf16_t)c[r];
        }
    }
}

// ===========================================================================
// WMMA kernel 4: split-K final GEMM: acc[b,j] += features[b,:] . out_W[:,j]
//   M=64 (4 tiles), N=144 (9 tiles), K=36096 = 24 chunks * 47 steps * 32
// ===========================================================================
#define KCHUNK_STEPS 47
__global__ void __launch_bounds__(32)
final_gemm_kernel(const bf16_t* __restrict__ F, const bf16_t* __restrict__ Wt,
                  float* __restrict__ acc) {
    int lane = threadIdx.x & 31;
    int hi = lane >> 4, lo = lane & 15;
    int m0 = blockIdx.x * 16, n0 = blockIdx.y * 16;
    int ks = blockIdx.z * (KCHUNK_STEPS * 32);
    const bf16_t* arow = F  + (size_t)(m0 + lo) * KOUT + hi * 16;
    const bf16_t* brow = Wt + (size_t)(n0 + lo) * KOUT + hi * 16;
    v8f c = {};
    for (int i = 0; i < KCHUNK_STEPS; ++i) {
        int k0 = ks + i * 32;
        v16bf a  = frag_a_bf(arow + k0);
        v16bf bf = frag_b_contig(brow + k0);
        c = WMMA_BF16(a, bf, c);
    }
#pragma unroll
    for (int r = 0; r < 8; ++r) {
        int m = m0 + r + hi * 8;
        atomicAdd(&acc[(size_t)m * LPAD + n0 + lo], c[r]);
    }
}

__global__ void final_out_kernel(const float* __restrict__ acc, const float* __restrict__ ob,
                                 float* __restrict__ out) {
    int idx = blockIdx.x * blockDim.x + threadIdx.x;
    if (idx >= BB * LL) return;
    int b = idx / LL, j = idx % LL;
    float v = acc[b * LPAD + j] + ob[j];
    out[idx] = 1.f / (1.f + expf(-v));
}

// ===========================================================================
// Host orchestration
// ===========================================================================
extern "C" void kernel_launch(void* const* d_in, const int* in_sizes, int n_in,
                              void* d_out, int out_size, void* d_ws, size_t ws_size,
                              hipStream_t stream) {
    (void)in_sizes; (void)n_in; (void)out_size; (void)ws_size;
    const int*   tok   = (const int*)d_in[0];
    const float* TH    = (const float*)d_in[1];
    const float* lab   = (const float*)d_in[2];
    const float* w12   = (const float*)d_in[3];
    const float* w23   = (const float*)d_in[4];
    const float* fre12 = (const float*)d_in[5];
    const float* fre23 = (const float*)d_in[6];
    const float* wi1_W = (const float*)d_in[7],  *wi1_b = (const float*)d_in[8];
    const float* wf1_W = (const float*)d_in[9],  *wf1_b = (const float*)d_in[10];
    const float* wo1_W = (const float*)d_in[11], *wo1_b = (const float*)d_in[12];
    const float* wu1_W = (const float*)d_in[13], *wu1_b = (const float*)d_in[14];
    const float* wi2_W = (const float*)d_in[15], *wi2_b = (const float*)d_in[16];
    const float* wf2_W = (const float*)d_in[17], *wf2_b = (const float*)d_in[18];
    const float* wo2_W = (const float*)d_in[19], *wo2_b = (const float*)d_in[20];
    const float* wu2_W = (const float*)d_in[21], *wu2_b = (const float*)d_in[22];
    const float* ui1_W = (const float*)d_in[23], *uf1_W = (const float*)d_in[24];
    const float* uo1_W = (const float*)d_in[25], *uu1_W = (const float*)d_in[26];
    const float* ui2_W = (const float*)d_in[27], *uf2_W = (const float*)d_in[28];
    const float* uo2_W = (const float*)d_in[29], *uu2_W = (const float*)d_in[30];
    const float* mix_W = (const float*)d_in[31], *mix_b = (const float*)d_in[32];
    const float* tt_W  = (const float*)d_in[33], *tt_b  = (const float*)d_in[34];
    const float* Amat  = (const float*)d_in[35];
    const float* Wp    = (const float*)d_in[36];
    const float* out_W = (const float*)d_in[37], *out_b = (const float*)d_in[38];
    float* out = (float*)d_out;

    const float* v1 = lab;
    const float* v2 = lab + NN1 * HH;
    const float* v3 = lab + (NN1 + NN2) * HH;

    // ---- workspace carve-up -------------------------------------------------
    char* base = (char*)d_ws; size_t off = 0;
    auto alloc = [&](size_t bytes) -> void* {
        void* p = base + off; off += (bytes + 255) & ~(size_t)255; return p;
    };
    auto f32buf = [&](size_t elems) -> float*  { return (float*)alloc(elems * 4); };
    auto bfbuf  = [&](size_t elems) -> bf16_t* { return (bf16_t*)alloc(elems * 2); };

    float *i11=f32buf(NN1*HH), *o11=f32buf(NN1*HH), *u11=f32buf(NN1*HH);
    float *h11=f32buf(NN1*HH), *c11=f32buf(NN1*HH);
    float *h21t=f32buf(NN2*HH), *cw21=f32buf(NN2*HH);
    float *i21=f32buf(NN2*HH), *f21=f32buf(NN2*HH), *o21=f32buf(NN2*HH), *u21=f32buf(NN2*HH);
    float *h21=f32buf(NN2*HH), *c21=f32buf(NN2*HH);
    float *h31t=f32buf(NN3*HH), *cw32=f32buf(NN3*HH);
    float *i31=f32buf(NN3*HH), *f31=f32buf(NN3*HH), *o31=f32buf(NN3*HH), *u31=f32buf(NN3*HH);
    float *h31=f32buf(NN3*HH);
    float *i32=f32buf(NN3*HH), *o32=f32buf(NN3*HH), *u32=f32buf(NN3*HH);
    float *h32=f32buf(NN3*HH), *c32=f32buf(NN3*HH);
    float *h22t=f32buf(NN2*HH), *vt2=f32buf(NN3*HH), *f22=f32buf(NN3*HH);
    float *t98=f32buf(NN3*HH), *addm22=f32buf(NN2*HH);
    float *i22=f32buf(NN2*HH), *o22=f32buf(NN2*HH), *u22=f32buf(NN2*HH);
    float *h22=f32buf(NN2*HH), *c22=f32buf(NN2*HH);
    float *h12t=f32buf(NN1*HH), *vt1=f32buf(NN2*HH), *f12=f32buf(NN2*HH);
    float *t34=f32buf(NN2*HH), *addm12=f32buf(NN1*HH);
    float *i12=f32buf(NN1*HH), *o12=f32buf(NN1*HH), *u12=f32buf(NN1*HH), *h12=f32buf(NN1*HH);
    float *t141=f32buf(LL*HH), *temp=f32buf(LL*HH);
    float *le_cat=f32buf(LL*768), *le=f32buf(LL*HH);
    bf16_t *le_bf   = bfbuf((size_t)LPAD*HH);
    bf16_t *ttw_t   = bfbuf((size_t)HH*768);
    bf16_t *text_bf = bfbuf((size_t)BB*SS*HH);
    bf16_t *text_tbf= bfbuf((size_t)BB*HH*SS);
    float  *logits  = f32buf((size_t)BB*LPAD*SS);
    bf16_t *att_bf  = bfbuf((size_t)BB*LPAD*SS);
    bf16_t *feat_bf = bfbuf((size_t)BB*KOUT);
    bf16_t *outw_t  = bfbuf((size_t)LPAD*KOUT);
    float  *acc     = f32buf(BB*LPAD);

    auto G = [](long n) { return dim3((unsigned)((n + 255) / 256)); };
    const dim3 T256(256);

    // ---- label-encoding LSTM chain (tiny, serialized on stream) ------------
    affine_act_kernel<<<G(NN1*HH),T256,0,stream>>>(v1,HH,wi1_W,wi1_b,nullptr,nullptr,i11,NN1,ACT_SIG);
    affine_act_kernel<<<G(NN1*HH),T256,0,stream>>>(v1,HH,wo1_W,wo1_b,nullptr,nullptr,o11,NN1,ACT_SIG);
    affine_act_kernel<<<G(NN1*HH),T256,0,stream>>>(v1,HH,wu1_W,wu1_b,nullptr,nullptr,u11,NN1,ACT_TANH);
    combine_kernel<<<G(NN1*HH),T256,0,stream>>>(i11,u11,o11,nullptr,nullptr,nullptr,h11,c11,NN1);
    mm_kernel<<<G(NN2*HH),T256,0,stream>>>(w12,h11,h21t,NN2,NN1,NN2,1);
    mm_kernel<<<G(NN2*HH),T256,0,stream>>>(w12,c11,cw21,NN2,NN1,NN2,1);
    affine_act_kernel<<<G(NN2*HH),T256,0,stream>>>(v2,HH,wi1_W,wi1_b,h21t,ui1_W,i21,NN2,ACT_SIG);
    affine_act_kernel<<<G(NN2*HH),T256,0,stream>>>(v2,HH,wf1_W,wf1_b,h21t,uf1_W,f21,NN2,ACT_SIG);
    affine_act_kernel<<<G(NN2*HH),T256,0,stream>>>(v2,HH,wo1_W,wo1_b,h21t,uo1_W,o21,NN2,ACT_SIG);
    affine_act_kernel<<<G(NN2*HH),T256,0,stream>>>(v2,HH,wu1_W,wu1_b,h21t,uu1_W,u21,NN2,ACT_TANH);
    combine_kernel<<<G(NN2*HH),T256,0,stream>>>(i21,u21,o21,f21,cw21,nullptr,h21,c21,NN2);
    mm_kernel<<<G(NN3*HH),T256,0,stream>>>(w23,h21,h31t,NN3,NN2,NN3,1);
    mm_kernel<<<G(NN3*HH),T256,0,stream>>>(w23,c21,cw32,NN3,NN2,NN3,1);
    affine_act_kernel<<<G(NN3*HH),T256,0,stream>>>(v3,HH,wi1_W,wi1_b,h31t,ui1_W,i31,NN3,ACT_SIG);
    affine_act_kernel<<<G(NN3*HH),T256,0,stream>>>(v3,HH,wf1_W,wf1_b,h31t,uf1_W,f31,NN3,ACT_SIG);
    affine_act_kernel<<<G(NN3*HH),T256,0,stream>>>(v3,HH,wo1_W,wo1_b,h31t,uo1_W,o31,NN3,ACT_SIG);
    affine_act_kernel<<<G(NN3*HH),T256,0,stream>>>(v3,HH,wu1_W,wu1_b,h31t,uu1_W,u31,NN3,ACT_TANH);
    combine_kernel<<<G(NN3*HH),T256,0,stream>>>(i31,u31,o31,f31,cw32,nullptr,h31,nullptr,NN3);
    affine_act_kernel<<<G(NN3*HH),T256,0,stream>>>(v3,HH,wi2_W,wi2_b,nullptr,nullptr,i32,NN3,ACT_SIG);
    affine_act_kernel<<<G(NN3*HH),T256,0,stream>>>(v3,HH,wo2_W,wo2_b,nullptr,nullptr,o32,NN3,ACT_SIG);
    affine_act_kernel<<<G(NN3*HH),T256,0,stream>>>(v3,HH,wu2_W,wu2_b,nullptr,nullptr,u32,NN3,ACT_TANH);
    combine_kernel<<<G(NN3*HH),T256,0,stream>>>(i32,u32,o32,nullptr,nullptr,nullptr,h32,c32,NN3);
    mm_kernel<<<G(NN2*HH),T256,0,stream>>>(fre23,h32,h22t,NN2,NN3,NN3,0);
    mm_kernel<<<G(NN3*HH),T256,0,stream>>>(w23,v2,vt2,NN3,NN2,NN3,1);
    affine_act_kernel<<<G(NN3*HH),T256,0,stream>>>(vt2,HH,wf2_W,wf2_b,h32,uf2_W,f22,NN3,ACT_SIG);
    emul_kernel<<<G(NN3*HH),T256,0,stream>>>(f22,c32,t98,NN3*HH);
    mm_kernel<<<G(NN2*HH),T256,0,stream>>>(w23,t98,addm22,NN2,NN3,NN3,0);
    affine_act_kernel<<<G(NN2*HH),T256,0,stream>>>(v2,HH,wi2_W,wi2_b,h22t,ui2_W,i22,NN2,ACT_SIG);
    affine_act_kernel<<<G(NN2*HH),T256,0,stream>>>(v2,HH,wo2_W,wo2_b,h22t,uo2_W,o22,NN2,ACT_SIG);
    affine_act_kernel<<<G(NN2*HH),T256,0,stream>>>(v2,HH,wu2_W,wu2_b,h22t,uu2_W,u22,NN2,ACT_TANH);
    combine_kernel<<<G(NN2*HH),T256,0,stream>>>(i22,u22,o22,nullptr,nullptr,addm22,h22,c22,NN2);
    mm_kernel<<<G(NN1*HH),T256,0,stream>>>(fre12,h22,h12t,NN1,NN2,NN2,0);
    mm_kernel<<<G(NN2*HH),T256,0,stream>>>(w12,v1,vt1,NN2,NN1,NN2,1);
    affine_act_kernel<<<G(NN2*HH),T256,0,stream>>>(vt1,HH,wf2_W,wf2_b,h22,uf2_W,f12,NN2,ACT_SIG);
    emul_kernel<<<G(NN2*HH),T256,0,stream>>>(f12,c22,t34,NN2*HH);
    mm_kernel<<<G(NN1*HH),T256,0,stream>>>(w12,t34,addm12,NN1,NN2,NN2,0);
    affine_act_kernel<<<G(NN1*HH),T256,0,stream>>>(v1,HH,wi2_W,wi2_b,h12t,ui2_W,i12,NN1,ACT_SIG);
    affine_act_kernel<<<G(NN1*HH),T256,0,stream>>>(v1,HH,wo2_W,wo2_b,h12t,uo2_W,o12,NN1,ACT_SIG);
    affine_act_kernel<<<G(NN1*HH),T256,0,stream>>>(v1,HH,wu2_W,wu2_b,h12t,uu2_W,u12,NN1,ACT_TANH);
    combine_kernel<<<G(NN1*HH),T256,0,stream>>>(i12,u12,o12,nullptr,nullptr,addm12,h12,nullptr,NN1);
    mm_kernel<<<G(LL*HH),T256,0,stream>>>(Amat,lab,t141,LL,LL,LL,0);
    affine_act_kernel<<<G(LL*HH),T256,0,stream>>>(t141,HH,Wp,nullptr,nullptr,nullptr,temp,LL,ACT_RELU);
    assemble_kernel<<<G(LL*768),T256,0,stream>>>(h11,h12,h21,h22,h31,h32,temp,le_cat);
    affine_act_kernel<<<G(LL*HH),T256,0,stream>>>(le_cat,768,mix_W,mix_b,nullptr,nullptr,le,LL,ACT_NONE);
    conv_le_kernel<<<G(LPAD*HH),T256,0,stream>>>(le,le_bf);

    // ---- heavy WMMA pipeline ----------------------------------------------
    convttw_kernel<<<G(768*HH),T256,0,stream>>>(tt_W,ttw_t);
    text_kernel<<<dim3(BB*SS/16),T256,0,stream>>>(TH,ttw_t,tt_b,text_bf,text_tbf);
    att_logits_kernel<<<dim3(BB,LPAD/16),T256,0,stream>>>(le_bf,text_bf,logits);
    softmax_kernel<<<dim3(BB,LPAD),dim3(128),0,stream>>>(logits,tok,att_bf);
    feat_kernel<<<dim3(BB,LPAD/16),T256,0,stream>>>(att_bf,text_tbf,feat_bf);
    convw_kernel<<<G((long)LPAD*KOUT),T256,0,stream>>>(out_W,outw_t);
    zero_kernel<<<G(BB*LPAD),T256,0,stream>>>(acc,BB*LPAD);
    final_gemm_kernel<<<dim3(BB/16,LPAD/16,24),dim3(32),0,stream>>>(feat_bf,outw_t,acc);
    final_out_kernel<<<G(BB*LL),T256,0,stream>>>(acc,out_b,out);
}
